// Model_24146306138603
// MI455X (gfx1250) — compile-verified, matrix operand order checked by reading, not run on hardware
//
#include <hip/hip_runtime.h>
#include <math.h>

typedef _Float16 f16;
typedef __attribute__((ext_vector_type(16))) _Float16 v16h;
typedef __attribute__((ext_vector_type(8)))  _Float16 h8;
typedef __attribute__((ext_vector_type(4)))  _Float16 h4;
typedef __attribute__((ext_vector_type(4)))  float    fl4;
typedef __attribute__((ext_vector_type(8)))  float    v8f;

#define BATCH    2
#define NHEADS   8
#define DHEAD    64
#define DMODEL   512
#define DFF      2048
#define ENC_L    2048
#define DEC_L    672
#define PRED_L   336
#define COUT     7
#define CIN      7

static inline int ceildiv(long a, long b) { return (int)((a + b - 1) / b); }

// ---------------------------------------------------------------------------
// weight conversion: f32 [K,N] row-major  ->  f16 [N,K] (column-major "BT")
// ---------------------------------------------------------------------------
__global__ void cvtT_f16_kernel(const float* __restrict__ in, f16* __restrict__ out,
                                int K, int N) {
    long i = (long)blockIdx.x * blockDim.x + threadIdx.x;   // indexes out[N][K]
    long total = (long)K * N;
    if (i >= total) return;
    int k = (int)(i % K);
    int n = (int)(i / K);
    out[i] = (f16)in[(long)k * N + n];
}

// ---------------------------------------------------------------------------
// WMMA GEMM: C[M,N] = act(scale * (A[M,K] x B[K,N]) + bias)
//   B supplied TRANSPOSED: BT[N][K] -> all tile loads are contiguous 16B.
//   AT = float (converted in the loader) or f16 (attention fragments).
//   FM x FN = 16x16 fragments per wave (4 waves, 2x2 grid per block).
//   GUARD=false: dims are tile multiples (true for every hot GEMM here) ->
//   register-staged, double-buffered, zero-branch software pipeline:
//     iter: [issue next-tile global loads] [WMMA on cur LDS buf]
//           [one loadcnt wait, convert+store to other buf] [one barrier]
//   GUARD=true: bounds-checked (final 512->7 projection only).
// ---------------------------------------------------------------------------
template <typename AT, int FM, int FN, bool GUARD>
__global__ __launch_bounds__(128)
void wmma_gemm(const AT* __restrict__ A, const f16* __restrict__ BT,
               const float* __restrict__ bias, float* __restrict__ C,
               int M, int N, int K,
               long strideA, long strideB, long strideC,
               float scale, int act)
{
    constexpr int TM  = 32 * FM;
    constexpr int TN  = 32 * FN;
    constexpr int LDK = 40;                      // halves: 80B stride (16B multiple)
    constexpr int NA4 = TM / 16;                 // fl4 per thread (f32 A)
    constexpr int NA8 = TM / 32;                 // h8  per thread (f16 A)
    constexpr int NB8 = TN / 32;                 // h8  per thread (B)
    __shared__ f16 As [2][TM][LDK];
    __shared__ f16 BsT[2][TN][LDK];
    A  += (long)blockIdx.z * strideA;
    BT += (long)blockIdx.z * strideB;
    C  += (long)blockIdx.z * strideC;
    const int tm   = blockIdx.y * TM;
    const int tn   = blockIdx.x * TN;
    const int tid  = threadIdx.x;
    const int wave = tid >> 5;
    const int lane = tid & 31;
    const int hi   = lane >> 4;                  // K sub-range select (ISA layout)
    const int l16  = lane & 15;
    const int wm   = (wave >> 1) * (16 * FM);
    const int wn   = (wave &  1) * (16 * FN);

    v8f acc[FM][FN] = {};
    fl4 stA4[NA4];                               // staging registers
    h8  stA8[NA8];
    h8  stB [NB8];

    auto load_tile = [&](int k0) {               // global -> regs (batched issue)
        if (sizeof(AT) == 4) {
            #pragma unroll
            for (int it = 0; it < NA4; ++it) {
                int id = tid + it * 128;
                int r = id >> 3, c4 = id & 7;
                stA4[it] = *(const fl4*)((const float*)A + (long)(tm + r) * K + k0 + c4 * 4);
            }
        } else {
            #pragma unroll
            for (int it = 0; it < NA8; ++it) {
                int id = tid + it * 128;
                int r = id >> 2, c8 = id & 3;
                stA8[it] = *(const h8*)((const f16*)A + (long)(tm + r) * K + k0 + c8 * 8);
            }
        }
        #pragma unroll
        for (int it = 0; it < NB8; ++it) {
            int id = tid + it * 128;
            int r = id >> 2, c8 = id & 3;
            stB[it] = *(const h8*)(BT + (long)(tn + r) * K + k0 + c8 * 8);
        }
        if (k0 + 32 < K) {                       // gfx1250 global_prefetch_b8 (+1 tile)
            int r = tid >> 3, c4 = tid & 7;
            __builtin_prefetch((const void*)((const AT*)A + (long)(tm + (r % TM)) * K + k0 + 32 + c4 * 4), 0, 3);
            int rb = tid >> 2, cb = tid & 3;
            __builtin_prefetch((const void*)(BT + (long)(tn + (rb % TN)) * K + k0 + 32 + cb * 8), 0, 3);
        }
    };
    auto store_tile = [&](int buf) {             // regs -> LDS (single loadcnt wait)
        if (sizeof(AT) == 4) {
            #pragma unroll
            for (int it = 0; it < NA4; ++it) {
                int id = tid + it * 128;
                int r = id >> 3, c4 = id & 7;
                h4 hv;
                hv[0] = (f16)stA4[it][0]; hv[1] = (f16)stA4[it][1];
                hv[2] = (f16)stA4[it][2]; hv[3] = (f16)stA4[it][3];
                *(h4*)&As[buf][r][c4 * 4] = hv;
            }
        } else {
            #pragma unroll
            for (int it = 0; it < NA8; ++it) {
                int id = tid + it * 128;
                int r = id >> 2, c8 = id & 3;
                *(h8*)&As[buf][r][c8 * 8] = stA8[it];
            }
        }
        #pragma unroll
        for (int it = 0; it < NB8; ++it) {
            int id = tid + it * 128;
            int r = id >> 2, c8 = id & 3;
            *(h8*)&BsT[buf][r][c8 * 8] = stB[it];
        }
    };
    auto compute = [&](int buf) {                // ds_load_b128 frags + WMMA
        v16h af[FM], bf[FN];
        #pragma unroll
        for (int fm = 0; fm < FM; ++fm) {
            const f16* row = &As[buf][wm + fm * 16 + l16][0];
            h8 lo = *(const h8*)(row + hi * 8);
            h8 hh = *(const h8*)(row + 16 + hi * 8);
            af[fm] = __builtin_shufflevector(lo, hh, 0,1,2,3,4,5,6,7,8,9,10,11,12,13,14,15);
        }
        #pragma unroll
        for (int fn = 0; fn < FN; ++fn) {
            const f16* row = &BsT[buf][wn + fn * 16 + l16][0];
            h8 lo = *(const h8*)(row + hi * 8);
            h8 hh = *(const h8*)(row + 16 + hi * 8);
            bf[fn] = __builtin_shufflevector(lo, hh, 0,1,2,3,4,5,6,7,8,9,10,11,12,13,14,15);
        }
        #pragma unroll
        for (int fm = 0; fm < FM; ++fm)
            #pragma unroll
            for (int fn = 0; fn < FN; ++fn)
                acc[fm][fn] = __builtin_amdgcn_wmma_f32_16x16x32_f16(
                    false, af[fm], false, bf[fn], (short)0, acc[fm][fn], false, false);
    };

    if (!GUARD) {
        load_tile(0);
        store_tile(0);
        __syncthreads();
        int buf = 0;
        for (int k0 = 0; k0 < K; k0 += 32) {
            bool has_next = (k0 + 32 < K);
            if (has_next) load_tile(k0 + 32);    // loads in flight during WMMA
            compute(buf);
            if (has_next) {
                store_tile(buf ^ 1);             // waits loads once, fills other buf
                __syncthreads();                 // one barrier per K-step
                buf ^= 1;
            }
        }
    } else {
        for (int k0 = 0; k0 < K; k0 += 32) {
            #pragma unroll
            for (int it = 0; it < TM / 4; ++it) {
                int idx = tid + it * 128;
                int r = idx >> 5, c = idx & 31;
                int ar = tm + r, ac = k0 + c;
                float av = (ar < M && ac < K) ? (float)A[(long)ar * K + ac] : 0.0f;
                As[0][r][c] = (f16)av;
            }
            #pragma unroll
            for (int it = 0; it < TN / 4; ++it) {
                int idx = tid + it * 128;
                int r = idx & 31, c = idx >> 5;  // r: k, c: n
                int br = k0 + r, bc = tn + c;
                BsT[0][c][r] = (br < K && bc < N) ? BT[(long)bc * K + br] : (f16)0.0f;
            }
            __syncthreads();
            compute(0);
            __syncthreads();
        }
    }

    #pragma unroll
    for (int fm = 0; fm < FM; ++fm) {
        #pragma unroll
        for (int fn = 0; fn < FN; ++fn) {
            #pragma unroll
            for (int r = 0; r < 8; ++r) {
                int row = tm + wm + fm * 16 + hi * 8 + r;    // C frag: VGPR r -> M=r+8*hi
                int col = tn + wn + fn * 16 + l16;
                if (!GUARD || (row < M && col < N)) {
                    float v = acc[fm][fn][r] * scale;
                    if (bias) v += bias[col];
                    if (act == 1) v = 0.5f * v * (1.0f + erff(v * 0.70710678118654752f));
                    C[(long)row * N + col] = v;
                }
            }
        }
    }
}

// ---------------------------------------------------------------------------
// Embedding: circular conv1d(k=3, no bias) + mark @ tmp_w + sinusoidal PE
// ---------------------------------------------------------------------------
__global__ void embed_kernel(const float* __restrict__ x, const float* __restrict__ mark,
                             const float* __restrict__ tokw, const float* __restrict__ tmpw,
                             float* __restrict__ y, int L)
{
    long i = (long)blockIdx.x * blockDim.x + threadIdx.x;
    long total = (long)BATCH * L * DMODEL;
    if (i >= total) return;
    int d = (int)(i & (DMODEL - 1));
    long bt = i >> 9;
    int t = (int)(bt % L);
    int b = (int)(bt / L);
    float acc = 0.0f;
    #pragma unroll
    for (int tap = 0; tap < 3; ++tap) {
        int src = t - 1 + tap;
        src = (src < 0) ? (L - 1) : ((src >= L) ? 0 : src);
        const float* xr = x + ((long)b * L + src) * CIN;
        const float* wr = tokw + (long)(tap * CIN) * DMODEL + d;
        #pragma unroll
        for (int c = 0; c < CIN; ++c) acc += xr[c] * wr[(long)c * DMODEL];
    }
    const float* mr = mark + ((long)b * L + t) * 5;
    #pragma unroll
    for (int m = 0; m < 5; ++m) acc += mr[m] * tmpw[m * DMODEL + d];
    int pi = d >> 1;
    float div = __expf((float)(2 * pi) * (-9.210340371976184f / (float)DMODEL));
    float ang = (float)t * div;
    acc += (d & 1) ? __cosf(ang) : __sinf(ang);
    y[i] = acc;
}

// ---------------------------------------------------------------------------
// LayerNorm (optionally with residual): y = LN(x + res) * g + b
// ---------------------------------------------------------------------------
__global__ __launch_bounds__(128)
void ln_kernel(const float* __restrict__ x, const float* __restrict__ res,
               const float* __restrict__ g, const float* __restrict__ b,
               float* __restrict__ y)
{
    __shared__ float red[128];
    long row = blockIdx.x;
    const float* xr = x + row * DMODEL;
    const float* rr = res ? res + row * DMODEL : nullptr;
    int tid = threadIdx.x;
    float s = 0.0f;
    for (int i = tid; i < DMODEL; i += 128) s += xr[i] + (rr ? rr[i] : 0.0f);
    red[tid] = s; __syncthreads();
    for (int o = 64; o > 0; o >>= 1) { if (tid < o) red[tid] += red[tid + o]; __syncthreads(); }
    float mu = red[0] / DMODEL; __syncthreads();
    float vs = 0.0f;
    for (int i = tid; i < DMODEL; i += 128) {
        float v = xr[i] + (rr ? rr[i] : 0.0f) - mu; vs += v * v;
    }
    red[tid] = vs; __syncthreads();
    for (int o = 64; o > 0; o >>= 1) { if (tid < o) red[tid] += red[tid + o]; __syncthreads(); }
    float rstd = rsqrtf(red[0] / DMODEL + 1e-5f);
    for (int i = tid; i < DMODEL; i += 128) {
        float v = xr[i] + (rr ? rr[i] : 0.0f);
        y[row * DMODEL + i] = (v - mu) * rstd * g[i] + b[i];
    }
}

// ---------------------------------------------------------------------------
// ProbSparse attention support kernels
// ---------------------------------------------------------------------------
__global__ void sample_score_kernel(const float* __restrict__ Q, const float* __restrict__ K,
                                    const int* __restrict__ samp, float* __restrict__ Mout,
                                    int LQ, int LK, int ns)
{
    long i = (long)blockIdx.x * blockDim.x + threadIdx.x;
    long total = (long)BATCH * NHEADS * LQ;
    if (i >= total) return;
    int l = (int)(i % LQ);
    int bh = (int)(i / LQ);
    int h = bh % NHEADS, b = bh / NHEADS;
    const float* qr = Q + ((long)b * LQ + l) * DMODEL + h * DHEAD;
    float mx = -3.0e38f, sm = 0.0f;
    for (int s = 0; s < ns; ++s) {
        int kk = samp[(long)l * ns + s];
        const float* kr = K + ((long)b * LK + kk) * DMODEL + h * DHEAD;
        float d = 0.0f;
        #pragma unroll
        for (int e = 0; e < DHEAD; ++e) d += qr[e] * kr[e];
        mx = fmaxf(mx, d); sm += d;
    }
    Mout[(long)bh * LQ + l] = mx - sm / (float)LK;
}

__global__ __launch_bounds__(256)
void topk_kernel(float* __restrict__ Mv, int* __restrict__ top, int LQ, int ntop)
{
    __shared__ float sv[256];
    __shared__ int   si[256];
    int bh = blockIdx.x;
    float* m = Mv + (long)bh * LQ;
    int tid = threadIdx.x;
    for (int t = 0; t < 32; ++t) {
        float bv = -3.0e38f; int bi = 0;
        if (t < ntop) {
            for (int i = tid; i < LQ; i += 256) {
                float v = m[i];
                if (v > bv) { bv = v; bi = i; }
            }
        }
        sv[tid] = bv; si[tid] = bi; __syncthreads();
        for (int o = 128; o > 0; o >>= 1) {
            if (tid < o && sv[tid + o] > sv[tid]) { sv[tid] = sv[tid + o]; si[tid] = si[tid + o]; }
            __syncthreads();
        }
        if (tid == 0) {
            top[(long)bh * 32 + t] = (t < ntop) ? si[0] : 0;
            if (t < ntop) m[si[0]] = -3.0e38f;
        }
        __syncthreads();
    }
}

__global__ void gather_qr_kernel(const float* __restrict__ Q, const int* __restrict__ top,
                                 f16* __restrict__ Qr, int LQ, int ntop)
{
    long i = (long)blockIdx.x * blockDim.x + threadIdx.x;
    long total = (long)BATCH * NHEADS * 32 * DHEAD;
    if (i >= total) return;
    int e = (int)(i & 63);
    long r = i >> 6;
    int u = (int)(r & 31);
    int bh = (int)(r >> 5);
    int h = bh % NHEADS, b = bh / NHEADS;
    float v = 0.0f;
    if (u < ntop) {
        int l = top[(long)bh * 32 + u];
        v = Q[((long)b * LQ + l) * DMODEL + h * DHEAD + e];
    }
    Qr[i] = (f16)v;
}

// [bh][LK][DHEAD] : per-head rows of K (BT layout for the scores GEMM, K=64)
__global__ void gather_ke_kernel(const float* __restrict__ K, f16* __restrict__ Kt, int LK)
{
    long i = (long)blockIdx.x * blockDim.x + threadIdx.x;
    long total = (long)BATCH * NHEADS * LK * DHEAD;
    if (i >= total) return;
    int e = (int)(i & 63);
    long r = i >> 6;
    int k = (int)(r % LK);
    int bh = (int)(r / LK);
    int h = bh % NHEADS, b = bh / NHEADS;
    Kt[i] = (f16)K[((long)b * LK + k) * DMODEL + h * DHEAD + e];
}

// [bh][DHEAD][LK] : V transposed (BT layout for the context-update GEMM, K=LK)
__global__ void gather_vt_kernel(const float* __restrict__ V, f16* __restrict__ Vt, int LK)
{
    long i = (long)blockIdx.x * blockDim.x + threadIdx.x;
    long total = (long)BATCH * NHEADS * DHEAD * LK;
    if (i >= total) return;
    int k = (int)(i % LK);
    long r = i / LK;
    int e = (int)(r & 63);
    int bh = (int)(r >> 6);
    int h = bh % NHEADS, b = bh / NHEADS;
    Vt[i] = (f16)V[((long)b * LK + k) * DMODEL + h * DHEAD + e];
}

__global__ __launch_bounds__(128)
void softmax_kernel(const float* __restrict__ scores, const int* __restrict__ top,
                    f16* __restrict__ probs, int LK, int causal)
{
    __shared__ float red[128];
    int row = blockIdx.x;                 // bh*32 + u
    const float* s = scores + (long)row * LK;
    f16* p = probs + (long)row * LK;
    int limit = causal ? top[row] : (LK - 1);
    int tid = threadIdx.x;
    float mx = -3.0e38f;
    for (int i = tid; i <= limit; i += 128) mx = fmaxf(mx, s[i]);
    red[tid] = mx; __syncthreads();
    for (int o = 64; o > 0; o >>= 1) { if (tid < o) red[tid] = fmaxf(red[tid], red[tid + o]); __syncthreads(); }
    mx = red[0]; __syncthreads();
    float sm = 0.0f;
    for (int i = tid; i <= limit; i += 128) sm += __expf(s[i] - mx);
    red[tid] = sm; __syncthreads();
    for (int o = 64; o > 0; o >>= 1) { if (tid < o) red[tid] += red[tid + o]; __syncthreads(); }
    float inv = 1.0f / red[0];
    for (int i = tid; i < LK; i += 128) {
        float v = (i <= limit) ? __expf(s[i] - mx) * inv : 0.0f;
        p[i] = (f16)v;
    }
}

__global__ void vmean_kernel(const float* __restrict__ V, float* __restrict__ vm, int LK)
{
    int i = blockIdx.x * blockDim.x + threadIdx.x;
    if (i >= BATCH * DMODEL) return;
    int d = i & (DMODEL - 1);
    int b = i >> 9;
    float s = 0.0f;
    for (int k = 0; k < LK; ++k) s += V[((long)b * LK + k) * DMODEL + d];
    vm[i] = s / (float)LK;
}

__global__ void fillctx_kernel(const float* __restrict__ vm, float* __restrict__ O, int LQ)
{
    long i = (long)blockIdx.x * blockDim.x + threadIdx.x;
    long total = (long)BATCH * LQ * DMODEL;
    if (i >= total) return;
    int d = (int)(i & (DMODEL - 1));
    long bt = i >> 9;
    int b = (int)(bt / LQ);
    O[i] = vm[b * DMODEL + d];
}

__global__ void cumsum_kernel(const float* __restrict__ V, float* __restrict__ O, int L)
{
    int i = blockIdx.x * blockDim.x + threadIdx.x;
    if (i >= BATCH * DMODEL) return;
    int d = i & (DMODEL - 1);
    int b = i >> 9;
    float s = 0.0f;
    for (int l = 0; l < L; ++l) {
        s += V[((long)b * L + l) * DMODEL + d];
        O[((long)b * L + l) * DMODEL + d] = s;
    }
}

__global__ void scatter_kernel(const float* __restrict__ upd, const int* __restrict__ top,
                               float* __restrict__ O, int LQ, int ntop)
{
    long i = (long)blockIdx.x * blockDim.x + threadIdx.x;
    long total = (long)BATCH * NHEADS * ntop * DHEAD;
    if (i >= total) return;
    int e = (int)(i & 63);
    long r = i >> 6;
    int u = (int)(r % ntop);
    int bh = (int)(r / ntop);
    int h = bh % NHEADS, b = bh / NHEADS;
    int l = top[(long)bh * 32 + u];
    O[((long)b * LQ + l) * DMODEL + h * DHEAD + e] = upd[((long)bh * 32 + u) * DHEAD + e];
}

// ---------------------------------------------------------------------------
// Distil: im2col (circular pad, K=3) and BN + ELU + maxpool(3, stride 2, pad 1)
// ---------------------------------------------------------------------------
__global__ void im2col_kernel(const float* __restrict__ x, f16* __restrict__ A, int L)
{
    long i = (long)blockIdx.x * blockDim.x + threadIdx.x;
    long total = (long)BATCH * L * (3 * DMODEL);
    if (i >= total) return;
    int c = (int)(i % (3 * DMODEL));
    long bt = i / (3 * DMODEL);
    int t = (int)(bt % L);
    int b = (int)(bt / L);
    int tap = c / DMODEL, ci = c % DMODEL;
    int src = t - 1 + tap;
    src = (src < 0) ? (L - 1) : ((src >= L) ? 0 : src);
    A[i] = (f16)x[((long)b * L + src) * DMODEL + ci];
}

__global__ void distil_pool_kernel(const float* __restrict__ y, const float* __restrict__ g,
                                   const float* __restrict__ bb, float* __restrict__ out, int L)
{
    long i = (long)blockIdx.x * blockDim.x + threadIdx.x;
    int Lo = L / 2;
    long total = (long)BATCH * Lo * DMODEL;
    if (i >= total) return;
    int d = (int)(i & (DMODEL - 1));
    long bt = i >> 9;
    int to = (int)(bt % Lo);
    int b = (int)(bt / Lo);
    const float sc = 0.9999950000374997f;  // 1/sqrt(1 + 1e-5)
    float mx = -3.0e38f;
    #pragma unroll
    for (int w = 0; w < 3; ++w) {
        int t = 2 * to - 1 + w;
        if (t < 0 || t >= L) continue;
        float v = g[d] * y[((long)b * L + t) * DMODEL + d] * sc + bb[d];
        v = (v > 0.0f) ? v : (__expf(v) - 1.0f);
        mx = fmaxf(mx, v);
    }
    out[i] = mx;
}

__global__ void slice_kernel(const float* __restrict__ P, float* __restrict__ out)
{
    int i = blockIdx.x * blockDim.x + threadIdx.x;
    int total = BATCH * PRED_L * COUT;
    if (i >= total) return;
    int c = i % COUT;
    int r = i / COUT;
    int t = r % PRED_L;
    int b = r / PRED_L;
    out[i] = P[((long)b * DEC_L + (DEC_L - PRED_L) + t) * COUT + c];
}

// ---------------------------------------------------------------------------
// Host orchestration
// ---------------------------------------------------------------------------
struct AttnW { const f16 *qw, *kw, *vw, *ow; const float *qb, *kb, *vb, *ob; };

extern "C" void kernel_launch(void* const* d_in, const int* in_sizes, int n_in,
                              void* d_out, int out_size, void* d_ws, size_t ws_size,
                              hipStream_t stream)
{
    (void)in_sizes; (void)n_in; (void)out_size; (void)ws_size;
    char* base = (char*)d_ws;
    size_t off = 0;
    auto alloc = [&](size_t bytes) -> void* {
        size_t a = (off + 255) & ~(size_t)255;
        void* r = base + a;
        off = a + bytes;
        return r;
    };
    // convert weight [K,N] -> f16 BT [N,K]
    auto cvtT = [&](const void* src, int K, int N) -> const f16* {
        long n = (long)K * N;
        f16* dst = (f16*)alloc((size_t)n * sizeof(f16));
        cvtT_f16_kernel<<<ceildiv(n, 256), 256, 0, stream>>>((const float*)src, dst, K, N);
        return dst;
    };

    // ---- input map (setup_inputs order) ----
    const float* x_enc      = (const float*)d_in[0];
    const float* x_mark_enc = (const float*)d_in[1];
    const float* x_dec      = (const float*)d_in[2];
    const float* x_mark_dec = (const float*)d_in[3];
    const float* enc_tok_w  = (const float*)d_in[4];
    const float* enc_tmp_w  = (const float*)d_in[5];
    const float* dec_tok_w  = (const float*)d_in[6];
    const float* dec_tmp_w  = (const float*)d_in[7];

    auto mkattn = [&](int i0) -> AttnW {
        AttnW w;
        w.qw = cvtT(d_in[i0 + 0], DMODEL, DMODEL); w.qb = (const float*)d_in[i0 + 1];
        w.kw = cvtT(d_in[i0 + 2], DMODEL, DMODEL); w.kb = (const float*)d_in[i0 + 3];
        w.vw = cvtT(d_in[i0 + 4], DMODEL, DMODEL); w.vb = (const float*)d_in[i0 + 5];
        w.ow = cvtT(d_in[i0 + 6], DMODEL, DMODEL); w.ob = (const float*)d_in[i0 + 7];
        return w;
    };

    AttnW enc0_attn = mkattn(8);
    const f16*   enc0_ff1w = cvtT(d_in[16], DMODEL, DFF);
    const float* enc0_ff1b = (const float*)d_in[17];
    const f16*   enc0_ff2w = cvtT(d_in[18], DFF, DMODEL);
    const float* enc0_ff2b = (const float*)d_in[19];
    const float *enc0_n1g = (const float*)d_in[20], *enc0_n1b = (const float*)d_in[21];
    const float *enc0_n2g = (const float*)d_in[22], *enc0_n2b = (const float*)d_in[23];

    AttnW enc1_attn = mkattn(24);
    const f16*   enc1_ff1w = cvtT(d_in[32], DMODEL, DFF);
    const float* enc1_ff1b = (const float*)d_in[33];
    const f16*   enc1_ff2w = cvtT(d_in[34], DFF, DMODEL);
    const float* enc1_ff2b = (const float*)d_in[35];
    const float *enc1_n1g = (const float*)d_in[36], *enc1_n1b = (const float*)d_in[37];
    const float *enc1_n2g = (const float*)d_in[38], *enc1_n2b = (const float*)d_in[39];

    const f16*   distil_w  = cvtT(d_in[40], 3 * DMODEL, DMODEL);   // BT [512][1536]
    const float* distil_b  = (const float*)d_in[41];
    const float* distil_bg = (const float*)d_in[42];
    const float* distil_bb = (const float*)d_in[43];
    const float *encN_g = (const float*)d_in[44], *encN_b = (const float*)d_in[45];

    AttnW dec_self  = mkattn(46);
    AttnW dec_cross = mkattn(54);
    const f16*   dec_ff1w = cvtT(d_in[62], DMODEL, DFF);
    const float* dec_ff1b = (const float*)d_in[63];
    const f16*   dec_ff2w = cvtT(d_in[64], DFF, DMODEL);
    const float* dec_ff2b = (const float*)d_in[65];
    const float *dec_n1g = (const float*)d_in[66], *dec_n1b = (const float*)d_in[67];
    const float *dec_n2g = (const float*)d_in[68], *dec_n2b = (const float*)d_in[69];
    const float *dec_n3g = (const float*)d_in[70], *dec_n3b = (const float*)d_in[71];
    const float *decN_g = (const float*)d_in[72], *decN_b = (const float*)d_in[73];
    const f16*   proj_w = cvtT(d_in[74], DMODEL, COUT);            // BT [7][512]
    const float* proj_b = (const float*)d_in[75];
    const int* samp_enc0  = (const int*)d_in[76];   // (2048, 32)
    const int* samp_enc1  = (const int*)d_in[77];   // (1024, 28)
    const int* samp_dself = (const int*)d_in[78];   // (672, 28)
    const int* samp_dcross= (const int*)d_in[79];   // (672, 28)

    // ---- activation / scratch buffers (sized for worst case, reused) ----
    const long MAXTOK = (long)BATCH * ENC_L;                 // 4096 rows
    float* X0  = (float*)alloc(MAXTOK * DMODEL * 4);
    float* X1  = (float*)alloc(MAXTOK * DMODEL * 4);
    float* T   = (float*)alloc(MAXTOK * DMODEL * 4);
    float* AO  = (float*)alloc(MAXTOK * DMODEL * 4);
    float* Qb  = (float*)alloc(MAXTOK * DMODEL * 4);
    float* Kb  = (float*)alloc(MAXTOK * DMODEL * 4);
    float* Vb  = (float*)alloc(MAXTOK * DMODEL * 4);
    float* Hf  = (float*)alloc(MAXTOK * DFF * 4);            // ffn hidden / conv out
    float* Eout= (float*)alloc((long)BATCH * (ENC_L/2) * DMODEL * 4);
    float* Mm  = (float*)alloc((long)BATCH * NHEADS * ENC_L * 4);
    int*   topb= (int*)  alloc((long)BATCH * NHEADS * 32 * 4);
    f16*  Qr16 = (f16*)  alloc((long)BATCH * NHEADS * 32 * DHEAD * 2);
    f16*  Ke16 = (f16*)  alloc((long)BATCH * NHEADS * ENC_L * DHEAD * 2);  // [bh][LK][64]
    f16*  Vt16 = (f16*)  alloc((long)BATCH * NHEADS * DHEAD * ENC_L * 2);  // [bh][64][LK]
    float* Sc  = (float*)alloc((long)BATCH * NHEADS * 32 * ENC_L * 4);
    f16*  Pr16 = (f16*)  alloc((long)BATCH * NHEADS * 32 * ENC_L * 2);
    float* Upd = (float*)alloc((long)BATCH * NHEADS * 32 * DHEAD * 4);
    float* Vm  = (float*)alloc((long)BATCH * DMODEL * 4);
    f16*  IC16 = (f16*)  alloc((long)BATCH * ENC_L * 3 * DMODEL * 2);
    float* Pout= (float*)alloc((long)BATCH * DEC_L * COUT * 4);

    // big GEMMs: 64x64 tile, 4 wmma/wave/K-step (all dims tile multiples)
    auto gemm = [&](const float* A, const f16* BT, const float* bias, float* C,
                    int M, int N, int K, float scale, int act) {
        dim3 g(ceildiv(N, 64), ceildiv(M, 64), 1);
        wmma_gemm<float, 2, 2, false><<<g, 128, 0, stream>>>(A, BT, bias, C, M, N, K, 0L, 0L, 0L, scale, act);
    };

    auto run_attention = [&](const float* Xq, int LQ, const float* Xkv, int LK,
                             const AttnW& w, const int* samp, int ns, int ntop,
                             int causal, float* Tout) {
        int Mq = BATCH * LQ, Mk = BATCH * LK;
        gemm(Xq,  w.qw, w.qb, Qb, Mq, DMODEL, DMODEL, 1.0f, 0);
        gemm(Xkv, w.kw, w.kb, Kb, Mk, DMODEL, DMODEL, 1.0f, 0);
        gemm(Xkv, w.vw, w.vb, Vb, Mk, DMODEL, DMODEL, 1.0f, 0);
        long tot = (long)BATCH * NHEADS * LQ;
        sample_score_kernel<<<ceildiv(tot, 256), 256, 0, stream>>>(Qb, Kb, samp, Mm, LQ, LK, ns);
        topk_kernel<<<BATCH * NHEADS, 256, 0, stream>>>(Mm, topb, LQ, ntop);
        tot = (long)BATCH * NHEADS * 32 * DHEAD;
        gather_qr_kernel<<<ceildiv(tot, 256), 256, 0, stream>>>(Qb, topb, Qr16, LQ, ntop);
        tot = (long)BATCH * NHEADS * LK * DHEAD;
        gather_ke_kernel<<<ceildiv(tot, 256), 256, 0, stream>>>(Kb, Ke16, LK);
        gather_vt_kernel<<<ceildiv(tot, 256), 256, 0, stream>>>(Vb, Vt16, LK);
        {   // scores[bh, 32, LK] = (1/8) Qr x K^T ; BT = Ke16 [LK][64]
            dim3 g(ceildiv(LK, 32), 1, BATCH * NHEADS);
            wmma_gemm<f16, 1, 1, false><<<g, 128, 0, stream>>>(Qr16, Ke16, nullptr, Sc, 32, LK, DHEAD,
                (long)32 * DHEAD, (long)LK * DHEAD, (long)32 * LK, 0.125f, 0);
        }
        softmax_kernel<<<BATCH * NHEADS * 32, 128, 0, stream>>>(Sc, topb, Pr16, LK, causal);
        {   // upd[bh, 32, 64] = probs x V ; BT = Vt16 [64][LK]
            dim3 g(1, 1, BATCH * NHEADS);
            wmma_gemm<f16, 1, 2, false><<<g, 128, 0, stream>>>(Pr16, Vt16, nullptr, Upd, 32, DHEAD, LK,
                (long)32 * LK, (long)DHEAD * LK, (long)32 * DHEAD, 1.0f, 0);
        }
        if (causal) {
            cumsum_kernel<<<ceildiv(BATCH * DMODEL, 256), 256, 0, stream>>>(Vb, AO, LQ);
        } else {
            vmean_kernel<<<ceildiv(BATCH * DMODEL, 256), 256, 0, stream>>>(Vb, Vm, LK);
            tot = (long)BATCH * LQ * DMODEL;
            fillctx_kernel<<<ceildiv(tot, 256), 256, 0, stream>>>(Vm, AO, LQ);
        }
        tot = (long)BATCH * NHEADS * ntop * DHEAD;
        scatter_kernel<<<ceildiv(tot, 256), 256, 0, stream>>>(Upd, topb, AO, LQ, ntop);
        gemm(AO, w.ow, w.ob, Tout, Mq, DMODEL, DMODEL, 1.0f, 0);
    };

    // =================== Encoder ===================
    float* cur = X0;
    float* alt = X1;
    long tot = (long)BATCH * ENC_L * DMODEL;
    embed_kernel<<<ceildiv(tot, 256), 256, 0, stream>>>(x_enc, x_mark_enc, enc_tok_w, enc_tmp_w, cur, ENC_L);

    // --- enc layer 0 (L=2048, u=32) ---
    run_attention(cur, ENC_L, cur, ENC_L, enc0_attn, samp_enc0, 32, 32, 0, T);
    ln_kernel<<<BATCH * ENC_L, 128, 0, stream>>>(cur, T, enc0_n1g, enc0_n1b, alt);
    { float* t = cur; cur = alt; alt = t; }
    gemm(cur, enc0_ff1w, enc0_ff1b, Hf, BATCH * ENC_L, DFF, DMODEL, 1.0f, 1);
    gemm(Hf,  enc0_ff2w, enc0_ff2b, T,  BATCH * ENC_L, DMODEL, DFF, 1.0f, 0);
    ln_kernel<<<BATCH * ENC_L, 128, 0, stream>>>(cur, T, enc0_n2g, enc0_n2b, alt);
    { float* t = cur; cur = alt; alt = t; }
    // --- distil (2048 -> 1024) ---
    tot = (long)BATCH * ENC_L * 3 * DMODEL;
    im2col_kernel<<<ceildiv(tot, 256), 256, 0, stream>>>(cur, IC16, ENC_L);
    {
        dim3 g(ceildiv(DMODEL, 64), ceildiv(BATCH * ENC_L, 64), 1);
        wmma_gemm<f16, 2, 2, false><<<g, 128, 0, stream>>>(IC16, distil_w, distil_b, Hf,
            BATCH * ENC_L, DMODEL, 3 * DMODEL, 0L, 0L, 0L, 1.0f, 0);
    }
    tot = (long)BATCH * (ENC_L / 2) * DMODEL;
    distil_pool_kernel<<<ceildiv(tot, 256), 256, 0, stream>>>(Hf, distil_bg, distil_bb, alt, ENC_L);
    { float* t = cur; cur = alt; alt = t; }
    const int L1 = ENC_L / 2;  // 1024

    // --- enc layer 1 (L=1024, u=28) ---
    run_attention(cur, L1, cur, L1, enc1_attn, samp_enc1, 28, 28, 0, T);
    ln_kernel<<<BATCH * L1, 128, 0, stream>>>(cur, T, enc1_n1g, enc1_n1b, alt);
    { float* t = cur; cur = alt; alt = t; }
    gemm(cur, enc1_ff1w, enc1_ff1b, Hf, BATCH * L1, DFF, DMODEL, 1.0f, 1);
    gemm(Hf,  enc1_ff2w, enc1_ff2b, T,  BATCH * L1, DMODEL, DFF, 1.0f, 0);
    ln_kernel<<<BATCH * L1, 128, 0, stream>>>(cur, T, enc1_n2g, enc1_n2b, alt);
    { float* t = cur; cur = alt; alt = t; }
    ln_kernel<<<BATCH * L1, 128, 0, stream>>>(cur, nullptr, encN_g, encN_b, Eout);

    // =================== Decoder ===================
    tot = (long)BATCH * DEC_L * DMODEL;
    embed_kernel<<<ceildiv(tot, 256), 256, 0, stream>>>(x_dec, x_mark_dec, dec_tok_w, dec_tmp_w, cur, DEC_L);

    run_attention(cur, DEC_L, cur, DEC_L, dec_self, samp_dself, 28, 28, 1, T);
    ln_kernel<<<BATCH * DEC_L, 128, 0, stream>>>(cur, T, dec_n1g, dec_n1b, alt);
    { float* t = cur; cur = alt; alt = t; }

    run_attention(cur, DEC_L, Eout, L1, dec_cross, samp_dcross, 28, 28, 0, T);
    ln_kernel<<<BATCH * DEC_L, 128, 0, stream>>>(cur, T, dec_n2g, dec_n2b, alt);
    { float* t = cur; cur = alt; alt = t; }

    gemm(cur, dec_ff1w, dec_ff1b, Hf, BATCH * DEC_L, DFF, DMODEL, 1.0f, 1);
    gemm(Hf,  dec_ff2w, dec_ff2b, T,  BATCH * DEC_L, DMODEL, DFF, 1.0f, 0);
    ln_kernel<<<BATCH * DEC_L, 128, 0, stream>>>(cur, T, dec_n3g, dec_n3b, alt);
    { float* t = cur; cur = alt; alt = t; }

    ln_kernel<<<BATCH * DEC_L, 128, 0, stream>>>(cur, nullptr, decN_g, decN_b, alt);
    { float* t = cur; cur = alt; alt = t; }

    // final projection 512 -> 7 (partial tile -> guarded 32x32 variant)
    {
        dim3 g(1, ceildiv(BATCH * DEC_L, 32), 1);
        wmma_gemm<float, 1, 1, true><<<g, 128, 0, stream>>>(cur, proj_w, proj_b, Pout,
            BATCH * DEC_L, COUT, DMODEL, 0L, 0L, 0L, 1.0f, 0);
    }
    slice_kernel<<<ceildiv((long)BATCH * PRED_L * COUT, 256), 256, 0, stream>>>(Pout, (float*)d_out);
}
